// SimpleImageMLP_WithDistLayer_89988154785962
// MI455X (gfx1250) — matrix operand order repaired and checked by reading, not compiled, not run on hardware
//
#include <hip/hip_runtime.h>
#include <hip/hip_bf16.h>

// ---------------------------------------------------------------------------
// Problem constants (from reference): B=1024, D_IN=1024, H=512, OUT=512
// ---------------------------------------------------------------------------
#define BDIM   1024
#define DIN    1024
#define HDIM   512
#define ODIM   512
#define EPS    1e-8f
#define MAXD   1000000.0f

typedef __bf16 bf16_t;
typedef bf16_t v16bf __attribute__((ext_vector_type(16)));
typedef float  v8f   __attribute__((ext_vector_type(8)));

// --- A fragment (16x32 bf16, "row striped across VGPRs within one lane") ---
// lanes 0-15 (khalf=0): elements 0..7 = K 0..7,  elements 8..15 = K 16..23
// lanes 16-31 (khalf=1): elements 0..7 = K 8..15, elements 8..15 = K 24..31
// slab = row-major f32 row pointer at current 32-wide K slab.
__device__ __forceinline__ v16bf load_fragA_bf16(const float* __restrict__ slab, int khalf) {
    const float* p = slab + 8 * khalf;
    float4 q0 = *(const float4*)(p + 0);
    float4 q1 = *(const float4*)(p + 4);
    float4 q2 = *(const float4*)(p + 16);
    float4 q3 = *(const float4*)(p + 20);
    v16bf v;
    v[0]  = (bf16_t)q0.x; v[1]  = (bf16_t)q0.y; v[2]  = (bf16_t)q0.z; v[3]  = (bf16_t)q0.w;
    v[4]  = (bf16_t)q1.x; v[5]  = (bf16_t)q1.y; v[6]  = (bf16_t)q1.z; v[7]  = (bf16_t)q1.w;
    v[8]  = (bf16_t)q2.x; v[9]  = (bf16_t)q2.y; v[10] = (bf16_t)q2.z; v[11] = (bf16_t)q2.w;
    v[12] = (bf16_t)q3.x; v[13] = (bf16_t)q3.y; v[14] = (bf16_t)q3.z; v[15] = (bf16_t)q3.w;
    return v;
}

// --- B fragment (32x16 bf16, "row striped across lanes within one VGPR") ---
// lane n, lanes 0-15: K 0..7, 16..23; lanes 16-31: K 8..15, 24..31.
// Since B[k][n] = W1[n][k], lane n reads two 8-float runs from W1 row n
// at k-offsets 8*khalf and 16 + 8*khalf.
__device__ __forceinline__ v16bf load_fragB_bf16(const float* __restrict__ slab, int khalf) {
    const float* p = slab + 8 * khalf;
    float4 q0 = *(const float4*)(p + 0);
    float4 q1 = *(const float4*)(p + 4);
    float4 q2 = *(const float4*)(p + 16);
    float4 q3 = *(const float4*)(p + 20);
    v16bf v;
    v[0]  = (bf16_t)q0.x; v[1]  = (bf16_t)q0.y; v[2]  = (bf16_t)q0.z; v[3]  = (bf16_t)q0.w;
    v[4]  = (bf16_t)q1.x; v[5]  = (bf16_t)q1.y; v[6]  = (bf16_t)q1.z; v[7]  = (bf16_t)q1.w;
    v[8]  = (bf16_t)q2.x; v[9]  = (bf16_t)q2.y; v[10] = (bf16_t)q2.z; v[11] = (bf16_t)q2.w;
    v[12] = (bf16_t)q3.x; v[13] = (bf16_t)q3.y; v[14] = (bf16_t)q3.z; v[15] = (bf16_t)q3.w;
    return v;
}

// ---------------------------------------------------------------------------
// Kernel 1: h = relu(x @ W1^T + bias)  via V_WMMA_F32_16X16X32_BF16
// Each wave computes a 32(M) x 32(N) patch: 2x2 tiles, 4 accumulators.
// Block = 128 threads (4 waves); waves split the N dimension.
// Grid: (BDIM/32, HDIM/128)
// ---------------------------------------------------------------------------
__global__ __launch_bounds__(128)
void gemm_bias_relu_wmma(const float* __restrict__ x,
                         const float* __restrict__ W1,
                         const float* __restrict__ bias,
                         float* __restrict__ h) {
    const int lane  = threadIdx.x & 31;
    const int wave  = threadIdx.x >> 5;
    const int m0    = blockIdx.x * 32;
    const int n0    = blockIdx.y * 128 + wave * 32;
    const int r     = lane & 15;   // row-in-tile for A, col-in-tile for B
    const int khalf = lane >> 4;

    const float* xr0 = x  + (size_t)(m0 + r)      * DIN;
    const float* xr1 = x  + (size_t)(m0 + 16 + r) * DIN;
    const float* wr0 = W1 + (size_t)(n0 + r)      * DIN;
    const float* wr1 = W1 + (size_t)(n0 + 16 + r) * DIN;

    v8f c00 = {}, c01 = {}, c10 = {}, c11 = {};

    for (int kk = 0; kk < DIN; kk += 32) {
        v16bf a0  = load_fragA_bf16(xr0 + kk, khalf);
        v16bf a1  = load_fragA_bf16(xr1 + kk, khalf);
        v16bf bm0 = load_fragB_bf16(wr0 + kk, khalf);
        v16bf bm1 = load_fragB_bf16(wr1 + kk, khalf);
        c00 = __builtin_amdgcn_wmma_f32_16x16x32_bf16(false, a0, false, bm0, (short)0, c00, false, false);
        c01 = __builtin_amdgcn_wmma_f32_16x16x32_bf16(false, a0, false, bm1, (short)0, c01, false, false);
        c10 = __builtin_amdgcn_wmma_f32_16x16x32_bf16(false, a1, false, bm0, (short)0, c10, false, false);
        c11 = __builtin_amdgcn_wmma_f32_16x16x32_bf16(false, a1, false, bm1, (short)0, c11, false, false);
    }

    // C/D layout: VGPR vr -> (lanes 0-15: M=vr, lanes 16-31: M=vr+8), N = lane&15
    const int ncol0 = n0 + r;
    const int ncol1 = n0 + 16 + r;
    const float bias0 = bias[ncol0];
    const float bias1 = bias[ncol1];
    #pragma unroll
    for (int vr = 0; vr < 8; ++vr) {
        const int mA = m0 + vr + 8 * khalf;
        const int mB = m0 + 16 + vr + 8 * khalf;
        h[(size_t)mA * HDIM + ncol0] = fmaxf(c00[vr] + bias0, 0.0f);
        h[(size_t)mA * HDIM + ncol1] = fmaxf(c01[vr] + bias1, 0.0f);
        h[(size_t)mB * HDIM + ncol0] = fmaxf(c10[vr] + bias0, 0.0f);
        h[(size_t)mB * HDIM + ncol1] = fmaxf(c11[vr] + bias1, 0.0f);
    }
}

// ---------------------------------------------------------------------------
// Kernel 2: Canberra distance + reciprocal similarity.
// 16x16 output tile per 256-thread block. LDS-staged h and Wd rows with a
// 4-float row pad (stride 516) so ds_load_b128 rotates banks per row.
// Grid: (BDIM/16, ODIM/16)
// ---------------------------------------------------------------------------
#define ROW_PITCH (HDIM + 4)   // 516 floats; 516*4 bytes is 16B-aligned

__global__ __launch_bounds__(256)
void canberra_sim(const float* __restrict__ h,
                  const float* __restrict__ Wd,
                  float* __restrict__ out) {
    extern __shared__ float smem[];
    float* hs = smem;                         // 16 rows * 516
    float* ws = smem + 16 * ROW_PITCH;        // 16 rows * 516

    const int tid = threadIdx.x;
    const int b0  = blockIdx.x * 16;
    const int o0  = blockIdx.y * 16;

    // Cooperative tile load: 16 rows x 128 float4 each, for both h and Wd.
    for (int i = tid; i < 16 * (HDIM / 4); i += 256) {
        const int row = i >> 7;          // / 128
        const int c4  = i & 127;
        float4 hv = *(const float4*)(h  + (size_t)(b0 + row) * HDIM + c4 * 4);
        float4 wv = *(const float4*)(Wd + (size_t)(o0 + row) * HDIM + c4 * 4);
        *(float4*)(hs + row * ROW_PITCH + c4 * 4) = hv;
        *(float4*)(ws + row * ROW_PITCH + c4 * 4) = wv;
    }
    __syncthreads();

    const int tb = tid & 15;   // b-row within tile
    const int to = tid >> 4;   // o-row within tile
    const float* hrow = hs + tb * ROW_PITCH;
    const float* wrow = ws + to * ROW_PITCH;

    float dist = 0.0f;
    #pragma unroll 4
    for (int k = 0; k < HDIM; k += 4) {
        float4 hv = *(const float4*)(hrow + k);
        float4 wv = *(const float4*)(wrow + k);
        {
            float num = fabsf(hv.x - wv.x);
            float den = fmaxf(fabsf(hv.x) + fabsf(wv.x), EPS);
            dist += num * __builtin_amdgcn_rcpf(den);
        }
        {
            float num = fabsf(hv.y - wv.y);
            float den = fmaxf(fabsf(hv.y) + fabsf(wv.y), EPS);
            dist += num * __builtin_amdgcn_rcpf(den);
        }
        {
            float num = fabsf(hv.z - wv.z);
            float den = fmaxf(fabsf(hv.z) + fabsf(wv.z), EPS);
            dist += num * __builtin_amdgcn_rcpf(den);
        }
        {
            float num = fabsf(hv.w - wv.w);
            float den = fmaxf(fabsf(hv.w) + fabsf(wv.w), EPS);
            dist += num * __builtin_amdgcn_rcpf(den);
        }
    }

    const float dc = fminf(fmaxf(dist + EPS, EPS), MAXD);
    out[(size_t)(b0 + tb) * ODIM + (o0 + to)] = 1.0f / dc;   // N_POW == 1.0
}

// ---------------------------------------------------------------------------
// Harness entry point
// ---------------------------------------------------------------------------
extern "C" void kernel_launch(void* const* d_in, const int* in_sizes, int n_in,
                              void* d_out, int out_size, void* d_ws, size_t ws_size,
                              hipStream_t stream) {
    (void)in_sizes; (void)n_in; (void)out_size; (void)ws_size;

    const float* x    = (const float*)d_in[0];   // (1024, 1024)
    const float* W1   = (const float*)d_in[1];   // (512, 1024)
    const float* bias = (const float*)d_in[2];   // (512,)
    const float* Wd   = (const float*)d_in[3];   // (512, 512)
    float*       out  = (float*)d_out;           // (1024, 512)
    float*       hbuf = (float*)d_ws;            // (1024, 512) scratch, 2 MB

    // Phase 1: h = relu(x @ W1^T + bias), bf16 WMMA with f32 accumulate.
    dim3 gGrid(BDIM / 32, HDIM / 128);
    gemm_bias_relu_wmma<<<gGrid, dim3(128), 0, stream>>>(x, W1, bias, hbuf);

    // Phase 2: Canberra distance -> similarity.
    dim3 cGrid(BDIM / 16, ODIM / 16);
    size_t ldsBytes = 2u * 16u * ROW_PITCH * sizeof(float);  // 66048 B
    canberra_sim<<<cGrid, dim3(256), ldsBytes, stream>>>(hbuf, Wd, out);
}
